// CausalSelfAttention_58153857188126
// MI455X (gfx1250) — matrix operand order repaired and checked by reading, not compiled, not run on hardware
//
#include <hip/hip_runtime.h>
#include <hip/hip_bf16.h>

// ---------------------------------------------------------------------------
// CausalSelfAttention for MI455X (gfx1250), wave32 WMMA f16 path.
// B=4, S=2048, D=1024, H=16, dh=64.
// Pipeline: cvt(x)->f16, LDS-tiled transpose+cvt(W)->f16 -> QKV WMMA GEMMs
//           -> flash attention (WMMA QK^T + WMMA P*V) -> proj WMMA GEMM.
// ---------------------------------------------------------------------------

typedef __attribute__((ext_vector_type(16))) _Float16 v16h;
typedef __attribute__((ext_vector_type(8)))  _Float16 v8h;
typedef __attribute__((ext_vector_type(8)))  float    v8f;

union AF { v16h v; _Float16 h[16]; v8h q[2]; };

constexpr int Bc  = 4;
constexpr int Sc  = 2048;
constexpr int Dc  = 1024;
constexpr int Hc  = 16;
constexpr int DHc = 64;
constexpr int BSc = Bc * Sc;                     // 8192 rows

static __device__ __forceinline__ v8f wmma_f16(const v16h& a, const v16h& b, const v8f& c) {
    return __builtin_amdgcn_wmma_f32_16x16x32_f16(false, a, false, b, (short)0, c, false, false);
}

// ---------------------------------------------------------------------------
// fp32 -> f16 elementwise convert
// ---------------------------------------------------------------------------
__global__ void cvt_f32_to_f16(const float* __restrict__ in, _Float16* __restrict__ out, int n) {
    int i = blockIdx.x * blockDim.x + threadIdx.x;
    if (i < n) out[i] = (_Float16)in[i];
}

// ---------------------------------------------------------------------------
// W [K=1024, N=1024] fp32 (row-major) -> Wt [N, K] f16 (row-major).
// LDS 32x32 tile transpose: coalesced reads AND coalesced (contiguous-K) writes.
// block = (32, 8); grid = (32, 32).
// ---------------------------------------------------------------------------
__global__ __launch_bounds__(256) void transpose_cvt(const float* __restrict__ in,
                                                     _Float16* __restrict__ out) {
    __shared__ float tile[32][33];
    const int nb = blockIdx.x * 32;   // n tile base (cols of in)
    const int kb = blockIdx.y * 32;   // k tile base (rows of in)
    const int tx = threadIdx.x;       // 0..31
    const int ty = threadIdx.y;       // 0..7
#pragma unroll
    for (int r = 0; r < 32; r += 8)
        tile[ty + r][tx] = in[(size_t)(kb + ty + r) * Dc + nb + tx];
    __syncthreads();
#pragma unroll
    for (int r = 0; r < 32; r += 8)
        out[(size_t)(nb + ty + r) * Dc + kb + tx] = (_Float16)tile[tx][ty + r];
}

// ---------------------------------------------------------------------------
// WMMA GEMM:  C[M=8192, N=1024] = A[M,K=1024] * Bt[N,K]^T + bias
// block = 128 threads (4 waves). Block tile 128x64; wave tile 32x64.
// mode 0: q   -> oh  f16 head layout [B,H,S,dh]
// mode 1: k   -> of  f32 present_k [B,H,S,dh]   + oh f16 same layout
// mode 2: v   -> of  f32 present_v [B,H,S,dh]   + oh f16 TRANSPOSED [B,H,dh,S]
//               (transposed copy staged through LDS for coalesced stores)
// mode 3: out -> of  f32 [BS, D]
// ---------------------------------------------------------------------------
__global__ __launch_bounds__(128) void gemm_wmma(
    const _Float16* __restrict__ A, const _Float16* __restrict__ Bt,
    const float* __restrict__ bias,
    _Float16* __restrict__ oh, float* __restrict__ of, int mode)
{
    // per-wave staging for the mode-2 transposed store: [dh=64][s=32(+pad)]
    __shared__ __align__(32) _Float16 vbuf[4][64][40];

    const int lane = threadIdx.x & 31;
    const int wave = threadIdx.x >> 5;
    const int lo = lane & 15;
    const int hi = lane >> 4;
    const int row0 = blockIdx.x * 128 + wave * 32;
    const int col0 = blockIdx.y * 64;

    v8f acc[2][4] = {};

    for (int k0 = 0; k0 < Dc; k0 += 32) {
        AF a[2];
#pragma unroll
        for (int i = 0; i < 2; ++i) {
            const _Float16* p = A + (size_t)(row0 + 16 * i + lo) * Dc + k0 + hi * 16;
            a[i].q[0] = *(const v8h*)(p);
            a[i].q[1] = *(const v8h*)(p + 8);
        }
        AF b[4];
#pragma unroll
        for (int j = 0; j < 4; ++j) {
            const _Float16* p = Bt + (size_t)(col0 + 16 * j + lo) * Dc + k0 + hi * 16;
            b[j].v = *(const v16h*)(p);
        }
#pragma unroll
        for (int i = 0; i < 2; ++i)
#pragma unroll
            for (int j = 0; j < 4; ++j)
                acc[i][j] = wmma_f16(a[i].v, b[j].v, acc[i][j]);
    }

#pragma unroll
    for (int i = 0; i < 2; ++i) {
#pragma unroll
        for (int j = 0; j < 4; ++j) {
            const int n = col0 + 16 * j + lo;
            const float bv = bias[n];
#pragma unroll
            for (int r = 0; r < 8; ++r) {
                const int m = row0 + 16 * i + 8 * hi + r;
                const float val = acc[i][j][r] + bv;
                const int b_ = m >> 11;          // m / S
                const int s_ = m & (Sc - 1);
                const int h_ = n >> 6;           // n / dh
                const int d_ = n & (DHc - 1);
                const size_t hidx = ((size_t)(b_ * Hc + h_) * Sc + s_) * DHc + d_;
                if (mode == 0) {
                    oh[hidx] = (_Float16)val;
                } else if (mode == 1) {
                    of[hidx] = val;
                    oh[hidx] = (_Float16)val;
                } else if (mode == 2) {
                    of[hidx] = val;
                    // stage transposed tile in LDS: row = dh (0..63), col = s_local
                    vbuf[wave][16 * j + lo][16 * i + 8 * hi + r] = (_Float16)val;
                } else {
                    of[(size_t)m * Dc + n] = val;
                }
            }
        }
    }

    if (mode == 2) {
        // wave tile = 32 s-rows (one batch) x 64 dh-cols (one head):
        // b,h fixed for the whole wave tile.
        const int b_ = row0 >> 11;
        const int s0 = row0 & (Sc - 1);
        const int h_ = col0 >> 6;
        const size_t base = ((size_t)(b_ * Hc + h_) * DHc) * Sc;
        // each lane emits 2 dh-rows of 32 contiguous halves (2 x 32B stores)
#pragma unroll
        for (int rr = 0; rr < 2; ++rr) {
            const int dhr = lane * 2 + rr;       // 0..63
            const v16h x0 = *(const v16h*)&vbuf[wave][dhr][0];
            const v16h x1 = *(const v16h*)&vbuf[wave][dhr][16];
            _Float16* dst = oh + base + (size_t)dhr * Sc + s0;
            *(v16h*)(dst)      = x0;
            *(v16h*)(dst + 16) = x1;
        }
    }
}

// ---------------------------------------------------------------------------
// Flash attention (causal). One wave = 32 query rows. Block = 4 waves = 128 q rows.
// grid = (S/128, B*H).  Q,K f16 [B,H,S,dh]; Vt f16 [B,H,dh,S]; O f16 [B,S,D].
// ---------------------------------------------------------------------------
__global__ __launch_bounds__(128) void attn_wmma(
    const _Float16* __restrict__ Qh, const _Float16* __restrict__ Kh,
    const _Float16* __restrict__ Vt, _Float16* __restrict__ O)
{
    __shared__ __align__(32) _Float16 pbuf[4][32][40];   // P staging, per wave

    const int lane = threadIdx.x & 31;
    const int wave = threadIdx.x >> 5;
    const int lo = lane & 15;
    const int hi = lane >> 4;
    const int bh = blockIdx.y;
    const int b_ = bh >> 4;
    const int h_ = bh & 15;
    const int q0 = blockIdx.x * 128 + wave * 32;

    const _Float16* Qp = Qh + (size_t)bh * Sc * DHc;
    const _Float16* Kp = Kh + (size_t)bh * Sc * DHc;
    const _Float16* Vp = Vt + (size_t)bh * DHc * Sc;

    // Q fragments: [m-tile][k-step of 32 over dh=64]
    AF qf[2][2];
#pragma unroll
    for (int i = 0; i < 2; ++i)
#pragma unroll
        for (int ds = 0; ds < 2; ++ds) {
            const _Float16* p = Qp + (size_t)(q0 + 16 * i + lo) * DHc + ds * 32 + hi * 16;
            qf[i][ds].q[0] = *(const v8h*)(p);
            qf[i][ds].q[1] = *(const v8h*)(p + 8);
        }

    v8f oacc[2][4] = {};
    float mrow[2][8], lrow[2][8];
#pragma unroll
    for (int i = 0; i < 2; ++i)
#pragma unroll
        for (int r = 0; r < 8; ++r) { mrow[i][r] = -3.0e38f; lrow[i][r] = 0.0f; }

    const int jEnd = q0 >> 5;
    for (int j = 0; j <= jEnd; ++j) {
        const int k0 = j * 32;

        // K fragments as B-operand of QK^T (row of K = column of K^T, contiguous dh)
        AF kb[2][2];
#pragma unroll
        for (int t = 0; t < 2; ++t)
#pragma unroll
            for (int ds = 0; ds < 2; ++ds)
                kb[t][ds].v = *(const v16h*)(Kp + (size_t)(k0 + 16 * t + lo) * DHc + ds * 32 + hi * 16);

        v8f sf[2][2] = {};
#pragma unroll
        for (int ds = 0; ds < 2; ++ds)
#pragma unroll
            for (int i = 0; i < 2; ++i)
#pragma unroll
                for (int t = 0; t < 2; ++t)
                    sf[i][t] = wmma_f16(qf[i][ds].v, kb[t][ds].v, sf[i][t]);

        // scale + causal mask (only the diagonal block needs masking)
#pragma unroll
        for (int i = 0; i < 2; ++i)
#pragma unroll
            for (int t = 0; t < 2; ++t)
#pragma unroll
                for (int r = 0; r < 8; ++r) {
                    float s = sf[i][t][r] * 0.125f;  // 1/sqrt(64)
                    if (j == jEnd && (k0 + 16 * t + lo) > (q0 + 16 * i + 8 * hi + r))
                        s = -3.0e38f;
                    sf[i][t][r] = s;
                }

        // streaming softmax update; row lives across the 16 lanes of a group
#pragma unroll
        for (int i = 0; i < 2; ++i)
#pragma unroll
            for (int r = 0; r < 8; ++r) {
                float rm = fmaxf(sf[i][0][r], sf[i][1][r]);
                rm = fmaxf(rm, __shfl_xor(rm, 1, 32));
                rm = fmaxf(rm, __shfl_xor(rm, 2, 32));
                rm = fmaxf(rm, __shfl_xor(rm, 4, 32));
                rm = fmaxf(rm, __shfl_xor(rm, 8, 32));
                const float mn = fmaxf(mrow[i][r], rm);
                const float corr = __expf(mrow[i][r] - mn);
                mrow[i][r] = mn;
                const float p0 = __expf(sf[i][0][r] - mn);
                const float p1 = __expf(sf[i][1][r] - mn);
                sf[i][0][r] = p0;
                sf[i][1][r] = p1;
                float rs = p0 + p1;
                rs += __shfl_xor(rs, 1, 32);
                rs += __shfl_xor(rs, 2, 32);
                rs += __shfl_xor(rs, 4, 32);
                rs += __shfl_xor(rs, 8, 32);
                lrow[i][r] = lrow[i][r] * corr + rs;
#pragma unroll
                for (int nt = 0; nt < 4; ++nt) oacc[i][nt][r] *= corr;
            }

        // stage P through LDS: C-frag layout -> A-frag layout
#pragma unroll
        for (int i = 0; i < 2; ++i)
#pragma unroll
            for (int t = 0; t < 2; ++t)
#pragma unroll
                for (int r = 0; r < 8; ++r)
                    pbuf[wave][16 * i + 8 * hi + r][16 * t + lo] = (_Float16)sf[i][t][r];

        AF pa[2];
#pragma unroll
        for (int i = 0; i < 2; ++i) {
            const _Float16* pp = &pbuf[wave][16 * i + lo][hi * 16];
            pa[i].q[0] = *(const v8h*)(pp);
            pa[i].q[1] = *(const v8h*)(pp + 8);
        }

        // V^T fragments (B-operand of P*V): lane lo = dh within tile, contiguous k
        AF vb[4];
#pragma unroll
        for (int nt = 0; nt < 4; ++nt)
            vb[nt].v = *(const v16h*)(Vp + (size_t)(nt * 16 + lo) * Sc + k0 + hi * 16);

#pragma unroll
        for (int i = 0; i < 2; ++i)
#pragma unroll
            for (int nt = 0; nt < 4; ++nt)
                oacc[i][nt] = wmma_f16(pa[i].v, vb[nt].v, oacc[i][nt]);
    }

    // normalize and write O in [B,S,D] layout (D index = h*64 + dh)
#pragma unroll
    for (int i = 0; i < 2; ++i)
#pragma unroll
        for (int r = 0; r < 8; ++r) {
            const float inv = 1.0f / lrow[i][r];
            const int s_ = q0 + 16 * i + 8 * hi + r;
#pragma unroll
            for (int nt = 0; nt < 4; ++nt)
                O[((size_t)(b_ * Sc + s_)) * Dc + h_ * DHc + nt * 16 + lo] =
                    (_Float16)(oacc[i][nt][r] * inv);
        }
}

// ---------------------------------------------------------------------------
// launcher
// ---------------------------------------------------------------------------
extern "C" void kernel_launch(void* const* d_in, const int* in_sizes, int n_in,
                              void* d_out, int out_size, void* d_ws, size_t ws_size,
                              hipStream_t stream) {
    (void)in_sizes; (void)n_in; (void)out_size; (void)ws_size;

    const float* x  = (const float*)d_in[0];
    const float* Wq = (const float*)d_in[1];
    const float* bq = (const float*)d_in[2];
    const float* Wk = (const float*)d_in[3];
    const float* bk = (const float*)d_in[4];
    const float* Wv = (const float*)d_in[5];
    const float* bv = (const float*)d_in[6];
    const float* Wp = (const float*)d_in[7];
    const float* bp = (const float*)d_in[8];

    constexpr size_t NXH   = (size_t)BSc * Dc;            // 8,388,608 halves
    constexpr size_t NW    = (size_t)Dc * Dc;             // 1,048,576 halves
    constexpr size_t NHEAD = (size_t)Bc * Hc * Sc * DHc;  // 8,388,608 halves

    _Float16* xh   = (_Float16*)d_ws;
    _Float16* wqt  = xh  + NXH;
    _Float16* wkt  = wqt + NW;
    _Float16* wvt  = wkt + NW;
    _Float16* wpt  = wvt + NW;
    _Float16* qh   = wpt + NW;
    _Float16* khd  = qh  + NHEAD;
    _Float16* vth  = khd + NHEAD;
    _Float16* ohlf = vth + NHEAD;                          // total ~88 MB

    float* outp  = (float*)d_out;
    float* presk = outp  + (size_t)BSc * Dc;
    float* presv = presk + NHEAD;

    const int nX = (int)NXH;
    cvt_f32_to_f16<<<(nX + 255) / 256, 256, 0, stream>>>(x, xh, nX);

    dim3 tb(32, 8);
    dim3 tg(Dc / 32, Dc / 32);     // (32, 32)
    transpose_cvt<<<tg, tb, 0, stream>>>(Wq, wqt);
    transpose_cvt<<<tg, tb, 0, stream>>>(Wk, wkt);
    transpose_cvt<<<tg, tb, 0, stream>>>(Wv, wvt);
    transpose_cvt<<<tg, tb, 0, stream>>>(Wp, wpt);

    dim3 gg(BSc / 128, Dc / 64);   // (64, 16)
    gemm_wmma<<<gg, 128, 0, stream>>>(xh, wqt, bq, qh,  nullptr, 0);
    gemm_wmma<<<gg, 128, 0, stream>>>(xh, wkt, bk, khd, presk,   1);
    gemm_wmma<<<gg, 128, 0, stream>>>(xh, wvt, bv, vth, presv,   2);

    dim3 ga(Sc / 128, Bc * Hc);    // (16, 64)
    attn_wmma<<<ga, 128, 0, stream>>>(qh, khd, vth, ohlf);

    gemm_wmma<<<gg, 128, 0, stream>>>(ohlf, wpt, bp, nullptr, outp, 3);
}
